// TransformerBlock_23605140259120
// MI455X (gfx1250) — matrix operand order repaired and checked by reading, not compiled, not run on hardware
//
#include <hip/hip_runtime.h>
#include <hip/hip_bf16.h>
#include <stdint.h>

typedef _Float16 half8  __attribute__((ext_vector_type(8)));
typedef _Float16 half16 __attribute__((ext_vector_type(16)));
typedef float    float8 __attribute__((ext_vector_type(8)));
typedef int      int4v  __attribute__((ext_vector_type(4)));

union H16 { half16 v; half8 h[2]; };

// ---------------------------------------------------------------------------
// Async global->LDS staging (CDNA5 GLOBAL_LOAD_ASYNC_TO_LDS_B128, ASYNCcnt),
// guarded: falls back to load+ds_store if the builtin is absent.
// Builtin prototype (probe-verified): (int4 AS1*, int4 AS3*, Ii, Ii).
// ---------------------------------------------------------------------------
#if defined(__has_builtin)
#if __has_builtin(__builtin_amdgcn_global_load_async_to_lds_b128)
#define HAVE_ASYNC_LDS 1
#endif
#endif
#ifndef HAVE_ASYNC_LDS
#define HAVE_ASYNC_LDS 0
#endif

typedef __attribute__((address_space(1))) int4v* g_int4p;   // global
typedef __attribute__((address_space(3))) int4v* l_int4p;   // LDS

__device__ inline void async_copy16(const _Float16* g, _Float16* l) {
#if HAVE_ASYNC_LDS
    __builtin_amdgcn_global_load_async_to_lds_b128(
        (g_int4p)(uintptr_t)g,
        (l_int4p)(uint32_t)(uintptr_t)l,   // generic LDS addr low 32 bits = LDS offset
        0, 0);
#else
    *(half8*)l = *(const half8*)g;
#endif
}

__device__ inline void wait_async0() {
#if HAVE_ASYNC_LDS
#if __has_builtin(__builtin_amdgcn_s_wait_asynccnt)
    __builtin_amdgcn_s_wait_asynccnt(0);
#else
    asm volatile("s_wait_asynccnt 0" ::: "memory");
#endif
#endif
}

// ---------------------------------------------------------------------------
// WMMA fragment loaders (V_WMMA_F32_16X16X32_F16 layouts, wave32)
// A: 16x32 (MxK). lane 0-15: M=lane, halves 0..7 -> K=hi*8+0..7 (hi=lane>>4),
//    halves 8..15 -> K=16+hi*8+0..7. Row-major source => two b128 loads.
// ---------------------------------------------------------------------------
__device__ inline half16 load_a_frag(const _Float16* A, int lda,
                                     int row0, int k0, int lane) {
    int m = lane & 15, hi = lane >> 4;
    const _Float16* p = A + (size_t)(row0 + m) * lda + k0;
    H16 u;
    u.h[0] = *(const half8*)(p + hi * 8);
    u.h[1] = *(const half8*)(p + 16 + hi * 8);
    return u.v;
}

// B: 32x16 (KxN). lane holds column N=lane&15; halves 0..7 -> K = hi*8+0..7,
// halves 8..15 -> K=16+hi*8+0..7. Source stored "transposed" (N rows, K contiguous, ld=ldb).
__device__ inline half16 load_b_frag(const _Float16* BT, int ldb,
                                     int n0, int k0, int lane) {
    int n = lane & 15, hi = lane >> 4;
    const _Float16* p = BT + (size_t)(n0 + n) * ldb + k0;
    H16 u;
    u.h[0] = *(const half8*)(p + hi * 8);
    u.h[1] = *(const half8*)(p + 16 + hi * 8);
    return u.v;
}

__device__ inline float8 wmma_f16(half16 a, half16 b, float8 c) {
    return __builtin_amdgcn_wmma_f32_16x16x32_f16(false, a, false, b,
                                                  (short)0, c, false, false);
}

// ---------------------------------------------------------------------------
// Weight prep: f32 [K][N] -> f16 [N][K] (transpose + convert), tiled via LDS
// ---------------------------------------------------------------------------
__global__ void transpose_f32_to_f16(const float* __restrict__ src,
                                     _Float16* __restrict__ dst, int K, int N) {
    __shared__ float tile[32][33];
    int kb = blockIdx.y * 32, nb = blockIdx.x * 32;
    int tx = threadIdx.x, ty = threadIdx.y; // 32 x 8
    for (int i = ty; i < 32; i += 8) {
        int k = kb + i, n = nb + tx;
        tile[i][tx] = (k < K && n < N) ? src[(size_t)k * N + n] : 0.f;
    }
    __syncthreads();
    for (int i = ty; i < 32; i += 8) {
        int n = nb + i, k = kb + tx;
        if (n < N && k < K) dst[(size_t)n * K + k] = (_Float16)tile[tx][i];
    }
}

// ---------------------------------------------------------------------------
// LayerNorm over rows of C=768, f32 in -> f16 out. One block per row.
// ---------------------------------------------------------------------------
__global__ void layernorm_f16(const float* __restrict__ x,
                              _Float16* __restrict__ out, int C) {
    int row = blockIdx.x;
    const float* p = x + (size_t)row * C;
    int lane = threadIdx.x & 31, wave = threadIdx.x >> 5;
    float s = 0.f, s2 = 0.f;
    for (int i = threadIdx.x; i < C; i += blockDim.x) {
        float v = p[i]; s += v; s2 += v * v;
    }
    #pragma unroll
    for (int off = 16; off > 0; off >>= 1) {
        s  += __shfl_xor(s,  off, 32);
        s2 += __shfl_xor(s2, off, 32);
    }
    __shared__ float red0[8], red1[8], stats[2];
    if (lane == 0) { red0[wave] = s; red1[wave] = s2; }
    __syncthreads();
    if (threadIdx.x == 0) {
        float ts = 0.f, ts2 = 0.f;
        int nw = blockDim.x >> 5;
        for (int i = 0; i < nw; ++i) { ts += red0[i]; ts2 += red1[i]; }
        float mu  = ts / C;
        float var = ts2 / C - mu * mu;
        stats[0] = mu; stats[1] = rsqrtf(var + 1e-5f);
    }
    __syncthreads();
    float mu = stats[0], rs = stats[1];
    for (int i = threadIdx.x; i < C; i += blockDim.x)
        out[(size_t)row * C + i] = (_Float16)((p[i] - mu) * rs);
}

// ---------------------------------------------------------------------------
// WMMA GEMM with double-buffered async LDS staging.
// out[M][N] = A[M][K](f16) * W, W stored as BT[N][K](f16).
// Block = 256 threads = 8 waves. Block tile: 128(M) x 64(N), K-step 64.
// LDS: A-tile 128x64 (stride 72 to spread banks), B-tile 64x64 (stride 72),
// both double-buffered. Next K-slab prefetched (async) during WMMA.
// MODE 0: f16 store; MODE 1: +bias + exact GELU, f16 store;
// MODE 2: +residual(f32), f32 store.  M%128==0, N%64==0, K%64==0.
// ---------------------------------------------------------------------------
#define LDA_SH 72
template <int MODE>
__global__ void gemm_wmma(const _Float16* __restrict__ A,
                          const _Float16* __restrict__ BT,
                          int M, int N, int K,
                          const float* __restrict__ bias,
                          const float* __restrict__ residual,
                          _Float16* __restrict__ outH,
                          float* __restrict__ outF) {
    __shared__ __align__(16) _Float16 shA[2][128 * LDA_SH];
    __shared__ __align__(16) _Float16 shB[2][64 * LDA_SH];

    int tid  = threadIdx.x;
    int lane = tid & 31, wave = tid >> 5;
    int rowB = blockIdx.x * 128;      // block row base
    int n0B  = blockIdx.y * 64;       // block col base

    // Stage one K-slab (64 wide) of A (128 rows) and B (64 rows) into LDS.
    auto stage = [&](int buf, int k0) {
        // A: 128 rows x 64 halves = 1024 x 16B chunks; 4 per thread
        #pragma unroll
        for (int i = 0; i < 4; ++i) {
            int c = tid + i * 256;
            int r = c >> 3, ck = (c & 7) * 8;
            async_copy16(A + (size_t)(rowB + r) * K + k0 + ck,
                         &shA[buf][r * LDA_SH + ck]);
        }
        // B: 64 rows x 64 halves = 512 x 16B chunks; 2 per thread
        #pragma unroll
        for (int i = 0; i < 2; ++i) {
            int c = tid + i * 256;
            int r = c >> 3, ck = (c & 7) * 8;
            async_copy16(BT + (size_t)(n0B + r) * K + k0 + ck,
                         &shB[buf][r * LDA_SH + ck]);
        }
    };

    stage(0, 0);
    wait_async0();
    __syncthreads();

    float8 acc[4] = {};
    for (int k0 = 0; k0 < K; k0 += 64) {
        int p = (k0 >> 6) & 1;
        if (k0 + 64 < K) stage(1 - p, k0 + 64);   // async prefetch next slab

        half16 a[2];
        #pragma unroll
        for (int kc = 0; kc < 2; ++kc)
            a[kc] = load_a_frag(&shA[p][0], LDA_SH, wave * 16, kc * 32, lane);
        #pragma unroll
        for (int j = 0; j < 4; ++j)
            #pragma unroll
            for (int kc = 0; kc < 2; ++kc) {
                half16 b = load_b_frag(&shB[p][0], LDA_SH, j * 16, kc * 32, lane);
                acc[j] = wmma_f16(a[kc], b, acc[j]);
            }

        wait_async0();       // next slab resident
        __syncthreads();     // and visible to / released by all waves
    }

    int m = lane & 15, hi = lane >> 4;
    int row0 = rowB + wave * 16;
    #pragma unroll
    for (int j = 0; j < 4; ++j) {
        int col = n0B + j * 16 + m;
        #pragma unroll
        for (int r = 0; r < 8; ++r) {
            int row = row0 + r + 8 * hi;
            float v = acc[j][r];
            if (MODE == 0) {
                outH[(size_t)row * N + col] = (_Float16)v;
            } else if (MODE == 1) {
                float xv = v + bias[col];
                float g = 0.5f * xv * (1.f + erff(xv * 0.70710678118f));
                outH[(size_t)row * N + col] = (_Float16)g;
            } else {
                outF[(size_t)row * N + col] = v + residual[(size_t)row * N + col];
            }
        }
    }
}

// ---------------------------------------------------------------------------
// Repack kqv [T][2304] f16 -> k,q: [12][T][64]; v: transposed [12][64][T]
// (reference reshape: c = three*768 + h*64 + d; three 0=k,1=q,2=v)
// ---------------------------------------------------------------------------
__global__ void repack_kqv(const _Float16* __restrict__ kqv,
                           _Float16* __restrict__ kb, _Float16* __restrict__ qb,
                           _Float16* __restrict__ vT, int T) {
    int idx = blockIdx.x * blockDim.x + threadIdx.x;
    int total = T * 2304;
    if (idx >= total) return;
    int t = idx / 2304, c = idx - t * 2304;
    int three = c / 768, rem = c - three * 768;
    int h = rem >> 6, d = rem & 63;
    _Float16 val = kqv[idx];
    if (three == 0)      kb[((size_t)h * T + t) * 64 + d] = val;
    else if (three == 1) qb[((size_t)h * T + t) * 64 + d] = val;
    else                 vT[((size_t)h * 64 + d) * T + t] = val;
}

// ---------------------------------------------------------------------------
// Flash attention with double-buffered async LDS staging of K/V tiles.
// Grid (T/64, 12 heads), block 128 = 4 waves; wave owns 16 query rows.
// Streams keys in blocks of 64; K-tile (64 keys x 64 d) and V-tile
// (64 d x 64 keys, from vT) staged to LDS (stride 72), shared by all waves,
// next key-block prefetched during WMMA+softmax of the current one.
// q,k: [h][T][64] f16; vT: [h][64][T] f16; out: [T][768] f16.
// Scale 1/8 folded into Q fragments.
// ---------------------------------------------------------------------------
__global__ void flash_attn(const _Float16* __restrict__ q,
                           const _Float16* __restrict__ k,
                           const _Float16* __restrict__ vT,
                           _Float16* __restrict__ out, int T) {
    constexpr int D = 64;
    int h = blockIdx.y;
    int tid = threadIdx.x;
    int lane = tid & 31, wave = tid >> 5;
    int q0 = blockIdx.x * 64 + wave * 16;
    int m = lane & 15, hi = lane >> 4;

    const _Float16* qh = q  + (size_t)h * T * D;
    const _Float16* kh = k  + (size_t)h * T * D;
    const _Float16* vh = vT + (size_t)h * D * T;

    __shared__ __align__(16) _Float16 shK[2][64 * LDA_SH];
    __shared__ __align__(16) _Float16 shV[2][64 * LDA_SH];
    __shared__ __align__(16) _Float16 pshm[4][16 * 64];
    _Float16* pbuf = pshm[wave];

    // Stage one key-block: K rows (key, d contiguous), V rows (d, key contiguous)
    auto stageKV = [&](int buf, int kb) {
        // 64 rows x 64 halves = 512 x 16B chunks each; 4 per thread each
        #pragma unroll
        for (int i = 0; i < 4; ++i) {
            int c = tid + i * 128;
            int r = c >> 3, ck = (c & 7) * 8;
            async_copy16(kh + (size_t)(kb + r) * D + ck, &shK[buf][r * LDA_SH + ck]);
        }
        #pragma unroll
        for (int i = 0; i < 4; ++i) {
            int c = tid + i * 128;
            int r = c >> 3, ck = (c & 7) * 8;
            async_copy16(vh + (size_t)r * T + kb + ck, &shV[buf][r * LDA_SH + ck]);
        }
    };

    // Q fragments with softmax scale folded in (0.125 exact in f16)
    half16 qf[2];
    qf[0] = load_a_frag(qh, D, q0, 0, lane);
    qf[1] = load_a_frag(qh, D, q0, 32, lane);
    #pragma unroll
    for (int e = 0; e < 16; ++e) {
        qf[0][e] = qf[0][e] * (_Float16)0.125f;
        qf[1][e] = qf[1][e] * (_Float16)0.125f;
    }

    float8 oacc[4] = {};
    float rm[8], rl[8];
    #pragma unroll
    for (int r = 0; r < 8; ++r) { rm[r] = -1e30f; rl[r] = 0.f; }

    stageKV(0, 0);
    wait_async0();
    __syncthreads();

    for (int kbase = 0; kbase < T; kbase += 64) {
        int p = (kbase >> 6) & 1;
        if (kbase + 64 < T) stageKV(1 - p, kbase + 64);  // async prefetch next block

        // S = Qs * K^T  (16 x 64 per wave), K fragments from LDS
        float8 s[4] = {};
        #pragma unroll
        for (int j = 0; j < 4; ++j) {
            half16 b0 = load_b_frag(&shK[p][0], LDA_SH, j * 16, 0, lane);
            s[j] = wmma_f16(qf[0], b0, s[j]);
            half16 b1 = load_b_frag(&shK[p][0], LDA_SH, j * 16, 32, lane);
            s[j] = wmma_f16(qf[1], b1, s[j]);
        }
        // per-row max (rows live in VGPR index r+8*hi; cols across 16 lanes)
        float nm[8];
        #pragma unroll
        for (int r = 0; r < 8; ++r)
            nm[r] = fmaxf(fmaxf(s[0][r], s[1][r]), fmaxf(s[2][r], s[3][r]));
        #pragma unroll
        for (int off = 1; off < 16; off <<= 1)
            #pragma unroll
            for (int r = 0; r < 8; ++r)
                nm[r] = fmaxf(nm[r], __shfl_xor(nm[r], off, 16));

        float alpha[8];
        #pragma unroll
        for (int r = 0; r < 8; ++r) {
            float newm = fmaxf(rm[r], nm[r]);
            alpha[r] = __expf(rm[r] - newm);
            rm[r] = newm;
        }
        // P = exp(S - m), row sums, stash P (C/D layout -> LDS tile)
        float rs[8] = {0.f, 0.f, 0.f, 0.f, 0.f, 0.f, 0.f, 0.f};
        #pragma unroll
        for (int j = 0; j < 4; ++j)
            #pragma unroll
            for (int r = 0; r < 8; ++r) {
                float pv = __expf(s[j][r] - rm[r]);
                rs[r] += pv;
                pbuf[(size_t)(r + 8 * hi) * 64 + j * 16 + m] = (_Float16)pv;
            }
        #pragma unroll
        for (int off = 1; off < 16; off <<= 1)
            #pragma unroll
            for (int r = 0; r < 8; ++r)
                rs[r] += __shfl_xor(rs[r], off, 16);
        #pragma unroll
        for (int r = 0; r < 8; ++r) rl[r] = rl[r] * alpha[r] + rs[r];
        #pragma unroll
        for (int j = 0; j < 4; ++j)
            #pragma unroll
            for (int r = 0; r < 8; ++r) oacc[j][r] *= alpha[r];

        // Re-read P as A-fragments (per-wave LDS region; in-order LDS ops)
        half16 pf[2];
        #pragma unroll
        for (int kc = 0; kc < 2; ++kc)
            pf[kc] = load_a_frag(pbuf, 64, 0, kc * 32, lane);

        // O += P * V, V fragments from LDS (rows = d, key contiguous)
        #pragma unroll
        for (int j = 0; j < 4; ++j)
            #pragma unroll
            for (int kc = 0; kc < 2; ++kc) {
                half16 vf = load_b_frag(&shV[p][0], LDA_SH, j * 16, kc * 32, lane);
                oacc[j] = wmma_f16(pf[kc], vf, oacc[j]);
            }

        wait_async0();       // next K/V block resident
        __syncthreads();     // visible to / released by all waves
    }
    // epilogue: divide by l, scatter to [T][768]
    #pragma unroll
    for (int j = 0; j < 4; ++j)
        #pragma unroll
        for (int r = 0; r < 8; ++r) {
            int row = q0 + r + 8 * hi;
            int col = h * 64 + j * 16 + m;
            out[(size_t)row * 768 + col] = (_Float16)(oacc[j][r] / rl[r]);
        }
}

// ---------------------------------------------------------------------------
extern "C" void kernel_launch(void* const* d_in, const int* in_sizes, int n_in,
                              void* d_out, int out_size, void* d_ws, size_t ws_size,
                              hipStream_t stream) {
    (void)in_sizes; (void)n_in; (void)out_size; (void)ws_size;
    const float* x       = (const float*)d_in[0]; // [4096][768]
    const float* W_kqv   = (const float*)d_in[1]; // [768][2304]
    const float* W_proj  = (const float*)d_in[2]; // [768][768]
    const float* W_fc    = (const float*)d_in[3]; // [768][1536]
    const float* b_fc    = (const float*)d_in[4]; // [1536]
    const float* W_cproj = (const float*)d_in[5]; // [1536][768]
    float* out = (float*)d_out;                   // [4096][768]

    const int T = 4096, C = 768, H = 12;

    char* ws = (char*)d_ws;
    size_t off = 0;
    auto alloc = [&](size_t bytes) -> void* {
        void* p = ws + off;
        off += (bytes + 255) & ~(size_t)255;
        return p;
    };
    _Float16* wkqvT   = (_Float16*)alloc((size_t)3 * C * C * 2);       // [2304][768]
    _Float16* wprojT  = (_Float16*)alloc((size_t)C * C * 2);           // [768][768]
    _Float16* wfcT    = (_Float16*)alloc((size_t)2 * C * C * 2);       // [1536][768]
    _Float16* wcprojT = (_Float16*)alloc((size_t)C * 2 * C * 2);       // [768][1536]
    _Float16* h1      = (_Float16*)alloc((size_t)T * C * 2);           // ln1 out
    _Float16* kqvH    = (_Float16*)alloc((size_t)T * 3 * C * 2);       // [T][2304]
    _Float16* kbuf    = (_Float16*)alloc((size_t)H * T * 64 * 2);
    _Float16* qbuf    = (_Float16*)alloc((size_t)H * T * 64 * 2);
    _Float16* vTbuf   = (_Float16*)alloc((size_t)H * 64 * T * 2);
    _Float16* attnH   = (_Float16*)alloc((size_t)T * C * 2);           // [T][768]
    float*    x1      = (float*)   alloc((size_t)T * C * 4);           // residual 1
    _Float16* h2      = (_Float16*)alloc((size_t)T * C * 2);           // ln2 out
    _Float16* a2      = (_Float16*)alloc((size_t)T * 2 * C * 2);       // gelu(fc)

    dim3 tb(32, 8);
    // Weight prep (transpose + f32->f16)
    transpose_f32_to_f16<<<dim3(3 * C / 32, C / 32), tb, 0, stream>>>(W_kqv, wkqvT, C, 3 * C);
    transpose_f32_to_f16<<<dim3(C / 32, C / 32), tb, 0, stream>>>(W_proj, wprojT, C, C);
    transpose_f32_to_f16<<<dim3(2 * C / 32, C / 32), tb, 0, stream>>>(W_fc, wfcT, C, 2 * C);
    transpose_f32_to_f16<<<dim3(C / 32, 2 * C / 32), tb, 0, stream>>>(W_cproj, wcprojT, 2 * C, C);

    // ln1
    layernorm_f16<<<T, 256, 0, stream>>>(x, h1, C);

    // kqv = h1 @ W_kqv  -> f16 [T][2304]
    gemm_wmma<0><<<dim3(T / 128, 3 * C / 64), 256, 0, stream>>>(
        h1, wkqvT, T, 3 * C, C, nullptr, nullptr, kqvH, nullptr);

    // split/reorder into q,k,[vT]
    {
        int total = T * 3 * C;
        repack_kqv<<<(total + 255) / 256, 256, 0, stream>>>(kqvH, kbuf, qbuf, vTbuf, T);
    }

    // attention (flash, streaming softmax)
    flash_attn<<<dim3(T / 64, H), 128, 0, stream>>>(qbuf, kbuf, vTbuf, attnH, T);

    // x1 = x + attn @ W_proj   (f32)
    gemm_wmma<2><<<dim3(T / 128, C / 64), 256, 0, stream>>>(
        attnH, wprojT, T, C, C, nullptr, x, nullptr, x1);

    // ln2
    layernorm_f16<<<T, 256, 0, stream>>>(x1, h2, C);

    // a2 = gelu(h2 @ W_fc + b_fc)  (f16)
    gemm_wmma<1><<<dim3(T / 128, 2 * C / 64), 256, 0, stream>>>(
        h2, wfcT, T, 2 * C, C, b_fc, nullptr, a2, nullptr);

    // out = x1 + a2 @ W_cproj  (f32)
    gemm_wmma<2><<<dim3(T / 128, C / 64), 256, 0, stream>>>(
        a2, wcprojT, T, C, 2 * C, nullptr, x1, nullptr, out);
}
